// EFLSTM_81561428951457
// MI455X (gfx1250) — matrix operand order repaired and checked by reading, not compiled, not run on hardware
//
#include <hip/hip_runtime.h>
#include <hip/hip_bf16.h>
#include <math.h>

// Problem constants (match reference)
#define B_    256
#define T_    256
#define DT    300
#define DA    74
#define DV    35
#define DD    409          // DT+DA+DV
#define H_    1024
#define G4    4096         // 4*H
#define KP    1440         // (H + DD) padded to multiple of 32: 1024+409 -> 1440
#define CELLN 512

// LDS staging geometry for the B (weight) panel
#define KCHUNK   96                     // K elements per chunk (3 WMMA k-steps)
#define NCHUNK   (KP / KCHUNK)          // 15 chunks
#define SLOTB    48                     // bytes per lane-slot (32 data + 16 pad)
#define FRAGB    (32 * SLOTB)           // 1536 B per (kk,gate) fragment region
#define CHUNKB   (12 * FRAGB)           // 18432 B per chunk (3 kk * 4 gates)

typedef __bf16 bf16;
typedef __attribute__((ext_vector_type(16))) __bf16 v16bf;
typedef __attribute__((ext_vector_type(8)))  __bf16 v8bf;
typedef __attribute__((ext_vector_type(8)))  float  v8f;

union V16 { v16bf v; v8bf h[2]; };
struct AFrag { V16 f[3]; };             // 3 k-steps of A fragments (24 VGPRs)

__device__ __forceinline__ float sigf(float x) { return 1.0f / (1.0f + __expf(-x)); }

__device__ __forceinline__ float xcat(const float* __restrict__ xt,
                                      const float* __restrict__ xa,
                                      const float* __restrict__ xv,
                                      int b, int t, int d) {
  if (d < DT)      return xt[(size_t)(b * T_ + t) * DT + d];
  if (d < DT + DA) return xa[(size_t)(b * T_ + t) * DA + (d - DT)];
  return xv[(size_t)(b * T_ + t) * DV + (d - DT - DA)];
}

// ---- One-time: pack W_cat[n][k] = [W_hh[n][:] | W_ih[n][:] | zero-pad] as bf16 ----
__global__ void prep_weights(const float* __restrict__ Wih, const float* __restrict__ Whh,
                             bf16* __restrict__ Wc) {
  int n = blockIdx.x;
  for (int k = threadIdx.x; k < KP; k += blockDim.x) {
    float v = 0.0f;
    if (k < H_)            v = Whh[(size_t)n * H_ + k];
    else if (k < H_ + DD)  v = Wih[(size_t)n * DD + (k - H_)];
    Wc[(size_t)n * KP + k] = (bf16)v;
  }
}

// ---- One-time: hx0 = [h=0 | x_0 | pad=0]; hx1 = 0 (pad cols stay 0 forever); c = 0 ----
__global__ void init_state(const float* __restrict__ xt, const float* __restrict__ xa,
                           const float* __restrict__ xv,
                           bf16* __restrict__ hx0, bf16* __restrict__ hx1,
                           float* __restrict__ c) {
  int b = blockIdx.x;
  for (int k = threadIdx.x; k < KP; k += blockDim.x) {
    float v = 0.0f;
    if (k >= H_ && k < H_ + DD) v = xcat(xt, xa, xv, b, 0, k - H_);
    hx0[(size_t)b * KP + k] = (bf16)v;
    hx1[(size_t)b * KP + k] = (bf16)0.0f;
  }
  for (int j = threadIdx.x; j < H_; j += blockDim.x) c[(size_t)b * H_ + j] = 0.0f;
}

// Per-wave A fragments for one chunk (rows m0+r, K = kbase..kbase+95).
__device__ __forceinline__ void aload(const bf16* __restrict__ arow, int kbase, int half,
                                      AFrag& a) {
#pragma unroll
  for (int kk = 0; kk < 3; kk++) {
    const v8bf* pa = (const v8bf*)(arow + kbase + kk * 32 + half * 16);
    a.f[kk].h[0] = pa[0];
    a.f[kk].h[1] = pa[1];   // +8 elements = +1 chunk
  }
}

// ---- Fused per-step kernel: gates = [h|x] @ Wcat^T (bf16 WMMA, f32 acc) + LSTM cell
//      B panel staged through double-buffered LDS; A via register pipeline.
// grid = (64, 2), block = 256 (8 waves). j0 = blockIdx.x*16 shared by all waves.
__global__ __launch_bounds__(256)
void lstm_step(const bf16* __restrict__ hx_r, const bf16* __restrict__ Wc,
               const float* __restrict__ bih, const float* __restrict__ bhh,
               float* __restrict__ c, float* __restrict__ h, bf16* __restrict__ hx_w,
               const float* __restrict__ xt, const float* __restrict__ xa,
               const float* __restrict__ xv, int t1) {
  __shared__ __align__(16) char smem[2 * CHUNKB];   // 36864 B, double-buffered B panel

  const int tid  = threadIdx.x;
  const int wave = tid >> 5;
  const int lane = tid & 31;
  const int half = lane >> 4;        // 0: lanes 0-15, 1: lanes 16-31
  const int r    = lane & 15;
  const int j0   = blockIdx.x * 16;                 // gate-column tile
  const int m0   = (blockIdx.y * 8 + wave) * 16;    // batch-row tile

  // ---------------- cooperative B staging setup ----------------
  // One chunk = 768 b128 data words; each thread owns words tid, tid+256, tid+512.
  // word w: q = w&1 (which 16B half of a 32B lane-slot), slot = (w>>1)&31,
  //         gk = w>>6 in [0,12): gg = gk&3 (gate), kk = gk>>2 (k-step).
  const bf16* wp[3];     // global source pointer for word i (chunk 0)
  int         lo[3];     // LDS byte offset for word i (within one chunk buffer)
#pragma unroll
  for (int i = 0; i < 3; i++) {
    int w    = tid + i * 256;
    int q    = w & 1;
    int slot = (w >> 1) & 31;
    int gk   = w >> 6;             // 0..11
    int gg   = gk & 3;
    int kk   = gk >> 2;
    int sr   = slot & 15;          // W-row within tile
    int sh   = slot >> 4;          // K half
    wp[i] = Wc + (size_t)(gg * H_ + j0 + sr) * KP + kk * 32 + sh * 16 + q * 8;
    lo[i] = gk * FRAGB + slot * SLOTB + q * 16;
  }

  const bf16* __restrict__ arow = hx_r + (size_t)(m0 + r) * KP;

  v8f ai = {}, af = {}, ag = {}, ao = {};

  // ---------------- prologue ----------------
  v8bf g0 = *(const v8bf*)(wp[0]);             // chunk 0 B words
  v8bf g1 = *(const v8bf*)(wp[1]);
  v8bf g2 = *(const v8bf*)(wp[2]);
  AFrag cur_a, nxt_a;
  aload(arow, 0, half, cur_a);                 // chunk 0 A frags

  *(v8bf*)(smem + lo[0]) = g0;                 // (implicit s_wait_loadcnt)
  *(v8bf*)(smem + lo[1]) = g1;
  *(v8bf*)(smem + lo[2]) = g2;
  g0 = *(const v8bf*)(wp[0] + KCHUNK);         // chunk 1 B words in flight
  g1 = *(const v8bf*)(wp[1] + KCHUNK);
  g2 = *(const v8bf*)(wp[2] + KCHUNK);
  __syncthreads();                             // buffer 0 ready

  const int so = lane * SLOTB;                 // this lane's slot offset

  // ---------------- main chunk loop ----------------
#pragma unroll 2
  for (int ci = 0; ci < NCHUNK; ci++) {
    const char* cbuf = smem + (ci & 1) * CHUNKB;
    char*       nbuf = smem + ((ci + 1) & 1) * CHUNKB;

    if (ci + 1 < NCHUNK) {
      *(v8bf*)(nbuf + lo[0]) = g0;             // stage chunk ci+1 (waits its loads)
      *(v8bf*)(nbuf + lo[1]) = g1;
      *(v8bf*)(nbuf + lo[2]) = g2;
    }
    if (ci + 2 < NCHUNK) {                     // launch chunk ci+2 B loads
      g0 = *(const v8bf*)(wp[0] + (ci + 2) * KCHUNK);
      g1 = *(const v8bf*)(wp[1] + (ci + 2) * KCHUNK);
      g2 = *(const v8bf*)(wp[2] + (ci + 2) * KCHUNK);
    }
    if (ci + 1 < NCHUNK) aload(arow, (ci + 1) * KCHUNK, half, nxt_a);

    // consume chunk ci: 3 k-steps x 4 gate WMMAs, B fragments from LDS
#pragma unroll
    for (int kk = 0; kk < 3; kk++) {
      const char* kb = cbuf + (kk * 4) * FRAGB;
      V16 fbi, fbf, fbg, fbo;
      fbi.h[0] = *(const v8bf*)(kb + 0 * FRAGB + so);
      fbi.h[1] = *(const v8bf*)(kb + 0 * FRAGB + so + 16);
      fbf.h[0] = *(const v8bf*)(kb + 1 * FRAGB + so);
      fbf.h[1] = *(const v8bf*)(kb + 1 * FRAGB + so + 16);
      fbg.h[0] = *(const v8bf*)(kb + 2 * FRAGB + so);
      fbg.h[1] = *(const v8bf*)(kb + 2 * FRAGB + so + 16);
      fbo.h[0] = *(const v8bf*)(kb + 3 * FRAGB + so);
      fbo.h[1] = *(const v8bf*)(kb + 3 * FRAGB + so + 16);
      ai = __builtin_amdgcn_wmma_f32_16x16x32_bf16(false, cur_a.f[kk].v, false, fbi.v, (short)0, ai, false, false);
      af = __builtin_amdgcn_wmma_f32_16x16x32_bf16(false, cur_a.f[kk].v, false, fbf.v, (short)0, af, false, false);
      ag = __builtin_amdgcn_wmma_f32_16x16x32_bf16(false, cur_a.f[kk].v, false, fbg.v, (short)0, ag, false, false);
      ao = __builtin_amdgcn_wmma_f32_16x16x32_bf16(false, cur_a.f[kk].v, false, fbo.v, (short)0, ao, false, false);
    }
    cur_a = nxt_a;
    __syncthreads();    // chunk ci reads done (buffer reusable) + chunk ci+1 stores visible
  }

  // ---------------- LSTM cell epilogue ----------------
  // C/D layout: VGPR v -> row m0+half*8+v, lane -> col j0+r.
  const int j = j0 + r;
  const float bias_i = bih[j]           + bhh[j];
  const float bias_f = bih[H_ + j]      + bhh[H_ + j];
  const float bias_g = bih[2 * H_ + j]  + bhh[2 * H_ + j];
  const float bias_o = bih[3 * H_ + j]  + bhh[3 * H_ + j];

#pragma unroll
  for (int v = 0; v < 8; v++) {
    const int m = m0 + half * 8 + v;
    const size_t idx = (size_t)m * H_ + j;
    float iv = sigf(ai[v] + bias_i);
    float fv = sigf(af[v] + bias_f);
    float gv = tanhf(ag[v] + bias_g);
    float ov = sigf(ao[v] + bias_o);
    float cv = fv * c[idx] + iv * gv;
    c[idx] = cv;
    float hv = ov * tanhf(cv);
    h[idx] = hv;                               // fp32 h for the MLP
    hx_w[(size_t)m * KP + j] = (bf16)hv;       // bf16 h for next step's A matrix
  }

  // Fused x-scatter: write x_{t1} into the write buffer (read next step).
  if (t1 < T_) {
    const int flat   = (blockIdx.y * gridDim.x + blockIdx.x) * blockDim.x + tid;
    const int stride = gridDim.x * gridDim.y * blockDim.x;   // 32768
    for (int idx2 = flat; idx2 < B_ * DD; idx2 += stride) {
      int b = idx2 / DD, d = idx2 % DD;
      hx_w[(size_t)b * KP + H_ + d] = (bf16)xcat(xt, xa, xv, b, t1, d);
    }
  }
}

// ---- MLP: hidden = relu(h @ W1^T + b1) ----
__global__ void mlp1(const float* __restrict__ h, const float* __restrict__ W1,
                     const float* __restrict__ b1, float* __restrict__ hid) {
  int id = blockIdx.x * blockDim.x + threadIdx.x;   // B_*CELLN = 131072 threads
  int b = id >> 9, j = id & 511;
  const float4* hv = (const float4*)(h + (size_t)b * H_);
  const float4* wv = (const float4*)(W1 + (size_t)j * H_);
  float s = 0.0f;
  for (int k = 0; k < H_ / 4; k++) {
    float4 a = hv[k], w = wv[k];
    s += a.x * w.x + a.y * w.y + a.z * w.z + a.w * w.w;
  }
  s += b1[j];
  hid[(size_t)b * CELLN + j] = fmaxf(s, 0.0f);
}

// ---- MLP: out = hidden @ W2^T + b2 (one wave per batch row) ----
__global__ void mlp2(const float* __restrict__ hid, const float* __restrict__ W2,
                     const float* __restrict__ b2, float* __restrict__ out) {
  int b = blockIdx.x;
  int lane = threadIdx.x;
  float s = 0.0f;
  for (int k = lane; k < CELLN; k += 32) s += hid[(size_t)b * CELLN + k] * W2[k];
  for (int off = 16; off > 0; off >>= 1) s += __shfl_xor(s, off, 32);
  if (lane == 0) out[b] = s + b2[0];
}

extern "C" void kernel_launch(void* const* d_in, const int* in_sizes, int n_in,
                              void* d_out, int out_size, void* d_ws, size_t ws_size,
                              hipStream_t stream) {
  const float* xt  = (const float*)d_in[0];
  const float* xa  = (const float*)d_in[1];
  const float* xv  = (const float*)d_in[2];
  const float* Wih = (const float*)d_in[3];
  const float* Whh = (const float*)d_in[4];
  const float* bih = (const float*)d_in[5];
  const float* bhh = (const float*)d_in[6];
  const float* W1  = (const float*)d_in[7];
  const float* b1  = (const float*)d_in[8];
  const float* W2  = (const float*)d_in[9];
  const float* b2  = (const float*)d_in[10];

  char* ws = (char*)d_ws;
  bf16*  Wc  = (bf16*)ws;                  ws += (size_t)G4 * KP * sizeof(bf16);   // 11.8 MB
  bf16*  hx0 = (bf16*)ws;                  ws += (size_t)B_ * KP * sizeof(bf16);   // 0.74 MB
  bf16*  hx1 = (bf16*)ws;                  ws += (size_t)B_ * KP * sizeof(bf16);   // 0.74 MB
  float* c   = (float*)ws;                 ws += (size_t)B_ * H_ * sizeof(float);  // 1 MB
  float* h   = (float*)ws;                 ws += (size_t)B_ * H_ * sizeof(float);  // 1 MB
  float* hid = (float*)ws;                 ws += (size_t)B_ * CELLN * sizeof(float);

  prep_weights<<<G4, 256, 0, stream>>>(Wih, Whh, Wc);
  init_state<<<B_, 256, 0, stream>>>(xt, xa, xv, hx0, hx1, c);

  for (int t = 0; t < T_; t++) {
    bf16* rd = (t & 1) ? hx1 : hx0;
    bf16* wr = (t & 1) ? hx0 : hx1;
    lstm_step<<<dim3(64, 2), 256, 0, stream>>>(rd, Wc, bih, bhh, c, h, wr,
                                               xt, xa, xv, t + 1);
  }

  mlp1<<<(B_ * CELLN) / 256, 256, 0, stream>>>(h, W1, b1, hid);
  mlp2<<<B_, 32, 0, stream>>>(hid, W2, b2, (float*)d_out);
}
